// SparseTF_38517266710962
// MI455X (gfx1250) — matrix operand order repaired and checked by reading, not compiled, not run on hardware
//
#include <hip/hip_runtime.h>
#include <hip/hip_bf16.h>
#include <math.h>

#define BATCH   4096
#define IN_DIM  8192
#define UNITS   4096
#define NNZ     2097152

#define BM 128
#define BN 128
#define BK 32
#define BKP (BK + 8)            // padded LDS pitch: 40 elems = 80B (16B multiple)
#define BUFB (BM * BKP * 2)     // bytes per LDS buffer = 10240; XOR-toggles 0<->10240

typedef __attribute__((ext_vector_type(16))) __bf16 v16bf;
typedef __attribute__((ext_vector_type(8)))  __bf16 v8bf;
typedef __attribute__((ext_vector_type(8)))  float  v8f;
typedef __attribute__((ext_vector_type(4)))  float  v4f;

union FragBF {
    v16bf v;
    v8bf  h[2];
};

// Branch-free tanh: copysign(1 - 2*rcp(exp2(2*log2e*|x|) + 1), x).
__device__ __forceinline__ float fast_tanh(float x) {
    const float ax = __builtin_fabsf(x);
    const float e  = __builtin_amdgcn_exp2f(ax * 2.885390081777927f); // 2*log2(e)
    const float r  = 1.0f - 2.0f * __builtin_amdgcn_rcpf(e + 1.0f);
    return __builtin_copysignf(r, x);
}

// One 16-byte async global->LDS copy (no VGPR data, tracked by ASYNCcnt).
__device__ __forceinline__ void async_copy_b128(unsigned lds_byte_addr, const void* gaddr) {
    asm volatile("global_load_async_to_lds_b128 %0, %1, off"
                 :: "v"(lds_byte_addr),
                    "v"((unsigned long long)(size_t)gaddr)
                 : "memory");
}

__device__ __forceinline__ void wait_async0() {
    asm volatile("s_wait_asynccnt 0x0" ::: "memory");
}

// ---------------------------------------------------------------------------
// 1) scatter-add sparse values into dense f32 weight [IN_DIM][UNITS]
// ---------------------------------------------------------------------------
__global__ void sparse_scatter_kernel(const float* __restrict__ vals,
                                      const int*   __restrict__ rows,
                                      const int*   __restrict__ cols,
                                      float*       __restrict__ dense) {
    int i = blockIdx.x * blockDim.x + threadIdx.x;
    if (i < NNZ) {
        atomicAdd(&dense[(size_t)rows[i] * UNITS + (size_t)cols[i]], vals[i]);
    }
}

// ---------------------------------------------------------------------------
// 2a) inputs f32 [BATCH][IN_DIM] -> bf16 (same layout)
// ---------------------------------------------------------------------------
__global__ __launch_bounds__(256)
void convert_a_kernel(const float* __restrict__ A, __bf16* __restrict__ Abf) {
    const size_t base = ((size_t)blockIdx.x * 256 + threadIdx.x) * 8;
    const v4f a0 = *(const v4f*)(A + base);
    const v4f a1 = *(const v4f*)(A + base + 4);
    v8bf b;
    #pragma unroll
    for (int j = 0; j < 4; ++j) { b[j] = (__bf16)a0[j]; b[j + 4] = (__bf16)a1[j]; }
    *(v8bf*)(Abf + base) = b;
}

// ---------------------------------------------------------------------------
// 2b) dense f32 [IN_DIM][UNITS] -> bf16 transposed [UNITS][IN_DIM]
// ---------------------------------------------------------------------------
__global__ __launch_bounds__(256)
void convert_bt_kernel(const float* __restrict__ dense, __bf16* __restrict__ Bbf) {
    __shared__ __bf16 tile[64][72];
    const int kt  = blockIdx.x * 64;
    const int nt  = blockIdx.y * 64;
    const int tid = threadIdx.x;

    const int nl = tid & 63;
    const int kl = tid >> 6;          // 0..3
    #pragma unroll
    for (int p = 0; p < 16; ++p) {
        const int k = kl + p * 4;
        tile[k][nl] = (__bf16)dense[(size_t)(kt + k) * UNITS + (nt + nl)];
    }
    __syncthreads();

    const int nr = tid >> 2;          // 0..63
    const int kc = (tid & 3) * 16;    // 0,16,32,48
    #pragma unroll
    for (int h = 0; h < 2; ++h) {
        v8bf v;
        #pragma unroll
        for (int j = 0; j < 8; ++j) v[j] = tile[kc + h * 8 + j][nr];
        *(v8bf*)&Bbf[(size_t)(nt + nr) * IN_DIM + kt + kc + h * 8] = v;
    }
}

// ---------------------------------------------------------------------------
// 3) C = tanh(A @ W + bias): bf16 WMMA 16x16x32, f32 accumulation.
//    Async global->LDS staging, double-buffered via XOR'd byte offset,
//    rolled K-loop with last iteration peeled (branch-free loop body).
// ---------------------------------------------------------------------------
__global__ __launch_bounds__(256)
void gemm_bias_tanh_kernel(const __bf16* __restrict__ Abf,   // [BATCH][IN_DIM]
                           const __bf16* __restrict__ Bbf,   // [UNITS][IN_DIM]
                           const float*  __restrict__ bias,
                           float*        __restrict__ C) {
    __shared__ alignas(16) __bf16 As[2][BM][BKP];
    __shared__ alignas(16) __bf16 Bs[2][BN][BKP];

    const int tid  = threadIdx.x;
    const int lane = tid & 31;
    const int wave = tid >> 5;
    const int wm   = wave & 3;        // rows wm*32..+32
    const int wn   = wave >> 2;       // cols wn*64..+64
    const int l16  = lane & 15;
    const int half = lane >> 4;

    const int m0 = blockIdx.y * BM;
    const int n0 = blockIdx.x * BN;

    // ---- staging: one 16B chunk per lane, 2 row-passes per operand
    const int s_kk  = (tid & 3) * 8;  // k-element offset of 16B chunk
    const int s_row = tid >> 2;       // 0..63

    const __bf16* ga0 = Abf + (size_t)(m0 + s_row)      * IN_DIM + s_kk;
    const __bf16* ga1 = Abf + (size_t)(m0 + s_row + 64) * IN_DIM + s_kk;
    const __bf16* gb0 = Bbf + (size_t)(n0 + s_row)      * IN_DIM + s_kk;
    const __bf16* gb1 = Bbf + (size_t)(n0 + s_row + 64) * IN_DIM + s_kk;

    const unsigned lA0 = (unsigned)(size_t)&As[0][s_row][s_kk];
    const unsigned lA1 = (unsigned)(size_t)&As[0][s_row + 64][s_kk];
    const unsigned lB0 = (unsigned)(size_t)&Bs[0][s_row][s_kk];
    const unsigned lB1 = (unsigned)(size_t)&Bs[0][s_row + 64][s_kk];

    // ---- loop-invariant fragment base addresses (buffer 0)
    const char* fA[2];
    #pragma unroll
    for (int mi = 0; mi < 2; ++mi)
        fA[mi] = (const char*)&As[0][wm * 32 + mi * 16 + l16][half * 8];
    const char* fB[4];
    #pragma unroll
    for (int ni = 0; ni < 4; ++ni)
        fB[ni] = (const char*)&Bs[0][wn * 64 + ni * 16 + l16][half * 8];

    v8f acc[2][4];
    #pragma unroll
    for (int mi = 0; mi < 2; ++mi)
        #pragma unroll
        for (int ni = 0; ni < 4; ++ni)
            #pragma unroll
            for (int r = 0; r < 8; ++r)
                acc[mi][ni][r] = 0.0f;

    // fragments per ISA 16-bit A/B layouts (two b128 LDS loads each) + 8 WMMAs
    auto compute_tile = [&](unsigned off) {
        FragBF af[2], bfr[4];
        #pragma unroll
        for (int mi = 0; mi < 2; ++mi) {
            af[mi].h[0] = *(const v8bf*)(fA[mi] + off);        // K {0..7 | 8..15}
            af[mi].h[1] = *(const v8bf*)(fA[mi] + off + 32);   // K {16..23 | 24..31}
        }
        #pragma unroll
        for (int ni = 0; ni < 4; ++ni) {
            bfr[ni].h[0] = *(const v8bf*)(fB[ni] + off);       // K {0..7 | 8..15}
            bfr[ni].h[1] = *(const v8bf*)(fB[ni] + off + 32);  // K {16..23 | 24..31}
        }
        #pragma unroll
        for (int mi = 0; mi < 2; ++mi)
            #pragma unroll
            for (int ni = 0; ni < 4; ++ni)
                acc[mi][ni] = __builtin_amdgcn_wmma_f32_16x16x32_bf16(
                    false, af[mi].v, false, bfr[ni].v,
                    (short)0, acc[mi][ni], false, false);
    };

    // ---- prologue: stage k=0 into buffer 0
    async_copy_b128(lA0, ga0);  async_copy_b128(lB0, gb0);
    async_copy_b128(lA1, ga1);  async_copy_b128(lB1, gb1);
    ga0 += BK; ga1 += BK; gb0 += BK; gb1 += BK;
    wait_async0();
    __syncthreads();

    const int NK = IN_DIM / BK;       // 256
    unsigned off = 0;                 // current compute buffer byte offset
    #pragma unroll 1
    for (int kb = 0; kb < NK - 1; ++kb) {
        // stage tile kb+1 (unconditional: branch-free loop body, no phi copies)
        const unsigned so = off ^ BUFB;
        async_copy_b128(lA0 + so, ga0);  async_copy_b128(lB0 + so, gb0);
        async_copy_b128(lA1 + so, ga1);  async_copy_b128(lB1 + so, gb1);
        ga0 += BK; ga1 += BK; gb0 += BK; gb1 += BK;

        compute_tile(off);

        wait_async0();    // tile kb+1 landed
        __syncthreads();  // all waves done reading current buffer
        off ^= BUFB;
    }
    compute_tile(off);    // peeled last tile (no staging)

    // ---- epilogue: bias + fast tanh (C layout: m = r + 8*half, n = l16)
    #pragma unroll
    for (int ni = 0; ni < 4; ++ni) {
        const int   n  = n0 + wn * 64 + ni * 16 + l16;
        const float bv = bias[n];
        #pragma unroll
        for (int mi = 0; mi < 2; ++mi) {
            const int mbase = m0 + wm * 32 + mi * 16 + half * 8;
            #pragma unroll
            for (int r = 0; r < 8; ++r) {
                C[(size_t)(mbase + r) * UNITS + n] = fast_tanh(acc[mi][ni][r] + bv);
            }
        }
    }
}

// ---------------------------------------------------------------------------
// Launch: memset dense -> scatter -> convert A -> convert/transpose W -> GEMM
// ws layout: [0,128M) dense f32 | [128M,192M) A bf16 | [192M,256M) Wt bf16
// ---------------------------------------------------------------------------
extern "C" void kernel_launch(void* const* d_in, const int* in_sizes, int n_in,
                              void* d_out, int out_size, void* d_ws, size_t ws_size,
                              hipStream_t stream) {
    const float* inputs = (const float*)d_in[0];
    const float* kvals  = (const float*)d_in[1];
    const float* bias   = (const float*)d_in[2];
    const int*   rows   = (const int*)d_in[3];
    const int*   cols   = (const int*)d_in[4];
    float*       out    = (float*)d_out;

    const size_t dense_bytes = (size_t)IN_DIM * UNITS * sizeof(float); // 128 MB
    float*  dense = (float*)d_ws;
    __bf16* Abf   = (__bf16*)((char*)d_ws + dense_bytes);                              // 64 MB
    __bf16* Bbf   = (__bf16*)((char*)d_ws + dense_bytes + (size_t)BATCH * IN_DIM * 2); // 64 MB

    hipMemsetAsync(dense, 0, dense_bytes, stream);

    sparse_scatter_kernel<<<NNZ / 256, 256, 0, stream>>>(kvals, rows, cols, dense);

    const int a_blocks = (int)(((size_t)BATCH * IN_DIM) / (256 * 8));
    convert_a_kernel<<<a_blocks, 256, 0, stream>>>(inputs, Abf);

    dim3 tgrid(IN_DIM / 64, UNITS / 64);
    convert_bt_kernel<<<tgrid, 256, 0, stream>>>(dense, Bbf);

    dim3 grid(UNITS / BN, BATCH / BM);
    gemm_bias_tanh_kernel<<<grid, 256, 0, stream>>>(Abf, Bbf, bias, out);
}
